// _MultiheadDiffAttn_25649544692110
// MI455X (gfx1250) — matrix operand order repaired and checked by reading, compiled, not run on hardware
//
#include <hip/hip_runtime.h>
#include <cstdint>
#include <cstddef>

typedef __bf16 bf16;
typedef __attribute__((ext_vector_type(8)))  bf16  v8bf;
typedef __attribute__((ext_vector_type(16))) bf16  v16bf;
typedef __attribute__((ext_vector_type(8)))  float v8f;

#define T_SEQ 2048
#define EMB   2048
#define NH    16
#define HD    64
#define LAMBDA_INIT  0.35550907f
#define ONE_MINUS_LI 0.64449093f
#define EPS 1e-5f

// ---------------------------------------------------------------- helpers

__device__ __forceinline__ v8f v8f_zero() {
  v8f z = {0.f,0.f,0.f,0.f,0.f,0.f,0.f,0.f};
  return z;
}

__device__ __forceinline__ v8f wmma_bf16(v16bf a, v16bf b, v8f c) {
  // D = A(16x32 bf16) * B(32x16 bf16) + C(16x16 f32)
  return __builtin_amdgcn_wmma_f32_16x16x32_bf16(
      /*neg_a=*/false, a, /*neg_b=*/false, b,
      /*c_mod=*/(short)0, c, /*reuse_a=*/false, /*reuse_b=*/false);
}

// A fragment (16x32 bf16): lane m = lane&15 picks the row; half = lane>>4
// selects K range: elems 0..15 -> K = half*16 + i
__device__ __forceinline__ v16bf load_a_frag(const bf16* row, int half) {
  v8bf lo = *(const v8bf*)(row + half * 16);
  v8bf hi = *(const v8bf*)(row + half * 16 + 8);
  v16bf a;
#pragma unroll
  for (int i = 0; i < 8; ++i) { a[i] = lo[i]; a[8 + i] = hi[i]; }
  return a;
}

// B fragment (32x16 bf16): lane n = lane&15 picks the column; lanes 0-15 hold
// K=0..15, lanes 16-31 hold K=16..31 -> 16 contiguous bf16 along K.
__device__ __forceinline__ v16bf load_bf16x16(const bf16* p) {
  v8bf lo = *(const v8bf*)p;
  v8bf hi = *(const v8bf*)(p + 8);
  v16bf b;
#pragma unroll
  for (int i = 0; i < 8; ++i) { b[i] = lo[i]; b[8 + i] = hi[i]; }
  return b;
}

// reductions across the 16 lanes of a half-wave (masks < 16 never cross halves)
__device__ __forceinline__ float half_max(float v) {
  v = fmaxf(v, __shfl_xor(v, 1, 32));
  v = fmaxf(v, __shfl_xor(v, 2, 32));
  v = fmaxf(v, __shfl_xor(v, 4, 32));
  v = fmaxf(v, __shfl_xor(v, 8, 32));
  return v;
}
__device__ __forceinline__ float half_sum(float v) {
  v += __shfl_xor(v, 1, 32);
  v += __shfl_xor(v, 2, 32);
  v += __shfl_xor(v, 4, 32);
  v += __shfl_xor(v, 8, 32);
  return v;
}

// ---------------------------------------------------------------- kernels

__global__ void cvt_f32_to_bf16(const float* __restrict__ in,
                                bf16* __restrict__ out, int n) {
  int i = blockIdx.x * blockDim.x + threadIdx.x;
  if (i < n) out[i] = (bf16)in[i];
}

// C[m][n] = sum_k A[m][k] * B[n][k]   (A: MxK bf16 row-major, B: NxK bf16 row-
// major == W so that C = A @ W.T).
// One wave -> 32(M) x 64(N) register tile: 2 A frags x 4 B frags = 8 WMMAs
// per 32-wide K step (12 b128 loads), maximizing WMMA density per operand set.
__global__ __launch_bounds__(256)
void gemm_bf16_nt(const bf16* __restrict__ A, const bf16* __restrict__ B,
                  float* __restrict__ C, int M, int N, int K) {
  const int lane = threadIdx.x & 31;
  const int wid  = threadIdx.x >> 5;
  const int half = lane >> 4;
  const int l15  = lane & 15;
  const int wm = wid >> 2, wn = wid & 3;
  const int m0 = blockIdx.y * 64 + wm * 32;   // two 16-row M fragments
  const int n0 = blockIdx.x * 256 + wn * 64;  // four 16-col N fragments

  v8f acc[2][4];
#pragma unroll
  for (int i = 0; i < 2; ++i)
#pragma unroll
    for (int j = 0; j < 4; ++j) acc[i][j] = v8f_zero();

  const bf16* arow0 = A + (size_t)(m0 + l15) * K;
  const bf16* arow1 = A + (size_t)(m0 + 16 + l15) * K;
  for (int k0 = 0; k0 < K; k0 += 32) {
    if (k0 + 64 < K) {
      __builtin_prefetch(arow0 + k0 + 64, 0, 3);  // near-scope streaming hint
      __builtin_prefetch(arow1 + k0 + 64, 0, 3);
    }
    v16bf a0 = load_a_frag(arow0 + k0, half);
    v16bf a1 = load_a_frag(arow1 + k0, half);
    v16bf b[4];
#pragma unroll
    for (int j = 0; j < 4; ++j)
      b[j] = load_bf16x16(B + (size_t)(n0 + j * 16 + l15) * K + k0 + half * 16);
#pragma unroll
    for (int j = 0; j < 4; ++j) acc[0][j] = wmma_bf16(a0, b[j], acc[0][j]);
#pragma unroll
    for (int j = 0; j < 4; ++j) acc[1][j] = wmma_bf16(a1, b[j], acc[1][j]);
  }
  // C/D layout: VGPR v -> M = v (+8 for lanes 16-31), N = lane&15
#pragma unroll
  for (int i = 0; i < 2; ++i)
#pragma unroll
    for (int j = 0; j < 4; ++j)
#pragma unroll
      for (int v = 0; v < 8; ++v)
        C[(size_t)(m0 + i * 16 + v + half * 8) * N + n0 + j * 16 + l15] =
            acc[i][j][v];
}

// in: [T][32][64] f32 -> rope -> out: [32][T][64] bf16 (head-major), *scale
__global__ void rope_cvt(const float* __restrict__ in,
                         const float* __restrict__ cs,
                         const float* __restrict__ sn,
                         bf16* __restrict__ out, float scale) {
  int idx = blockIdx.x * blockDim.x + threadIdx.x;
  if (idx >= T_SEQ * 32 * HD) return;
  int d  = idx & 63;
  int h2 = (idx >> 6) & 31;
  int t  = idx >> 11;
  float x  = in[idx];
  float ot = (d < 32) ? -in[idx + 32] : in[idx - 32];
  float r  = (x * cs[t * HD + d] + ot * sn[t * HD + d]) * scale;
  out[((size_t)h2 * T_SEQ + t) * HD + d] = (bf16)r;
}

// in: [T][16][128] f32 -> out: [16][128][T] bf16 (transposed per head)
__global__ void v_cvt_tr(const float* __restrict__ in, bf16* __restrict__ out) {
  int idx = blockIdx.x * blockDim.x + threadIdx.x;
  if (idx >= NH * 128 * T_SEQ) return;
  int t  = idx & (T_SEQ - 1);
  int dv = (idx >> 11) & 127;
  int h  = idx >> 18;
  out[idx] = (bf16)in[((size_t)t * NH + h) * 128 + dv];
}

__global__ void lambda_k(const float* lq1, const float* lk1,
                         const float* lq2, const float* lk2, float* lam) {
  if (threadIdx.x == 0 && blockIdx.x == 0) {
    float s1 = 0.f, s2 = 0.f;
    for (int i = 0; i < HD; ++i) { s1 += lq1[i] * lk1[i]; s2 += lq2[i] * lk2[i]; }
    *lam = __expf(s1) - __expf(s2) + LAMBDA_INIT;
  }
}

// Differential flash attention. Grid: (T/64, NH); block 256 = 8 waves.
// Wave w: query tile = w>>1 (16 rows), differential half = w&1.
__global__ __launch_bounds__(256)
void diff_flash(const bf16* __restrict__ Q,   // [32][T][64]
                const bf16* __restrict__ K,   // [32][T][64]
                const bf16* __restrict__ Vt,  // [16][128][T]
                const float* __restrict__ lam_ptr,
                const float* __restrict__ subln,
                bf16* __restrict__ O) {       // [T][2048]
  __shared__ bf16  pstage[8][16][32];    // per-wave P staging (8 KB)
  __shared__ float ocomb[4][16][128];    // half-0 outputs per qtile (32 KB)

  const int lane  = threadIdx.x & 31;
  const int wid   = threadIdx.x >> 5;
  const int half  = lane >> 4;
  const int l15   = lane & 15;
  const int qtile = wid >> 1;
  const int ph    = wid & 1;
  const int h     = blockIdx.y;
  const int h2    = 2 * h + ph;
  const int m0    = blockIdx.x * 64 + qtile * 16;

  // Q fragments for d=0..31 and d=32..63 (loaded once)
  const bf16* qrow = Q + ((size_t)h2 * T_SEQ + (m0 + l15)) * HD;
  v16bf aq0 = load_a_frag(qrow, half);
  v16bf aq1 = load_a_frag(qrow + 32, half);

  float mrun[8], lrun[8];
  v8f o[8];
#pragma unroll
  for (int v = 0; v < 8; ++v) { mrun[v] = -1e30f; lrun[v] = 0.f; }
#pragma unroll
  for (int t = 0; t < 8; ++t) o[t] = v8f_zero();

  const int s_last = m0 + 15;               // causal bound (T % 32 == 0)
  for (int s0 = 0; s0 <= s_last; s0 += 32) {
    // ---- scores S = Q @ K^T for two 16-key sub-blocks
    v8f sf[2];
#pragma unroll
    for (int nb = 0; nb < 2; ++nb) {
      const int scol = s0 + nb * 16 + l15;
      const bf16* krow = K + ((size_t)h2 * T_SEQ + scol) * HD;
      v8f acc = v8f_zero();
      acc = wmma_bf16(aq0, load_bf16x16(krow + half * 16), acc);
      acc = wmma_bf16(aq1, load_bf16x16(krow + 32 + half * 16), acc);
      sf[nb] = acc;
    }
    // ---- causal mask + online softmax update (per C-fragment row v)
    float alpha[8];
#pragma unroll
    for (int v = 0; v < 8; ++v) {
      const int row = m0 + v + half * 8;
      const int c0 = s0 + l15, c1 = s0 + 16 + l15;
      float x0 = (c0 <= row) ? sf[0][v] : -1e30f;
      float x1 = (c1 <= row) ? sf[1][v] : -1e30f;
      float bm = half_max(fmaxf(x0, x1));
      float mnew = fmaxf(mrun[v], bm);
      float p0 = __expf(x0 - mnew);
      float p1 = __expf(x1 - mnew);
      float rs = half_sum(p0 + p1);
      alpha[v] = __expf(mrun[v] - mnew);
      lrun[v]  = lrun[v] * alpha[v] + rs;
      mrun[v]  = mnew;
      pstage[wid][v + half * 8][l15]      = (bf16)p0;
      pstage[wid][v + half * 8][16 + l15] = (bf16)p1;
    }
#pragma unroll
    for (int t = 0; t < 8; ++t)
#pragma unroll
      for (int v = 0; v < 8; ++v) o[t][v] *= alpha[v];

    // wave-local LDS RAW: wait for DS stores before fragment reload
    asm volatile("s_wait_dscnt 0x0" ::: "memory");
    v16bf pa = load_a_frag(&pstage[wid][l15][0], half);

    // ---- O += P @ V  (8 dim-tiles of 16)
#pragma unroll
    for (int t = 0; t < 8; ++t) {
      const bf16* vp =
          Vt + ((size_t)h * 128 + t * 16 + l15) * T_SEQ + s0 + half * 16;
      o[t] = wmma_bf16(pa, load_bf16x16(vp), o[t]);
    }
  }

  // normalize each half independently (softmax denominator)
#pragma unroll
  for (int t = 0; t < 8; ++t)
#pragma unroll
    for (int v = 0; v < 8; ++v) o[t][v] /= lrun[v];

  if (ph == 0) {
#pragma unroll
    for (int t = 0; t < 8; ++t)
#pragma unroll
      for (int v = 0; v < 8; ++v)
        ocomb[qtile][v + half * 8][t * 16 + l15] = o[t][v];
  }
  __syncthreads();
  if (ph == 1) {
    const float lam = *lam_ptr;
    float ssq[8] = {0.f,0.f,0.f,0.f,0.f,0.f,0.f,0.f};
#pragma unroll
    for (int t = 0; t < 8; ++t)
#pragma unroll
      for (int v = 0; v < 8; ++v) {
        float val = ocomb[qtile][v + half * 8][t * 16 + l15] - lam * o[t][v];
        o[t][v] = val;
        ssq[v] += val * val;
      }
    float rowscale[8];
#pragma unroll
    for (int v = 0; v < 8; ++v) {
      float s = half_sum(ssq[v]);
      rowscale[v] = ONE_MINUS_LI / sqrtf(s * (1.f / 128.f) + EPS);
    }
#pragma unroll
    for (int t = 0; t < 8; ++t) {
      const int n = t * 16 + l15;
      const float sc = subln[n];
#pragma unroll
      for (int v = 0; v < 8; ++v)
        O[(size_t)(m0 + v + half * 8) * EMB + h * 128 + n] =
            (bf16)(o[t][v] * rowscale[v] * sc);
    }
  }
}

// ---------------------------------------------------------------- driver

extern "C" void kernel_launch(void* const* d_in, const int* in_sizes, int n_in,
                              void* d_out, int out_size, void* d_ws,
                              size_t ws_size, hipStream_t stream) {
  (void)in_sizes; (void)n_in; (void)out_size; (void)ws_size;
  const float* x    = (const float*)d_in[0];
  const float* cosv = (const float*)d_in[1];
  const float* sinv = (const float*)d_in[2];
  const float* Wq   = (const float*)d_in[3];
  const float* Wk   = (const float*)d_in[4];
  const float* Wv   = (const float*)d_in[5];
  const float* Wo   = (const float*)d_in[6];
  const float* lq1  = (const float*)d_in[7];
  const float* lk1  = (const float*)d_in[8];
  const float* lq2  = (const float*)d_in[9];
  const float* lk2  = (const float*)d_in[10];
  const float* subl = (const float*)d_in[11];
  float* out = (float*)d_out;

  char* w = (char*)d_ws;
  size_t off = 0;
  auto take = [&](size_t bytes) -> char* {
    char* p = w + off;
    off += (bytes + 255) & ~(size_t)255;
    return p;
  };
  const size_t NEL = (size_t)T_SEQ * EMB;      // 4 M elements
  bf16*  xbf    = (bf16*)take(NEL * 2);
  bf16*  wqbf   = (bf16*)take(NEL * 2);
  bf16*  wkbf   = (bf16*)take(NEL * 2);
  bf16*  wvbf   = (bf16*)take(NEL * 2);
  bf16*  wobf   = (bf16*)take(NEL * 2);
  float* qf     = (float*)take(NEL * 4);
  float* kf     = (float*)take(NEL * 4);
  float* vf     = (float*)take(NEL * 4);
  bf16*  qbf    = (bf16*)take(NEL * 2);
  bf16*  kbf    = (bf16*)take(NEL * 2);
  bf16*  vtbf   = (bf16*)take(NEL * 2);
  bf16*  attnbf = (bf16*)take(NEL * 2);
  float* lam    = (float*)take(256);

  const int n   = (int)NEL;
  const int blk = 256;
  const int gEl = (n + blk - 1) / blk;
  const dim3 gGemm(EMB / 256, T_SEQ / 64);

  // 1) convert inputs/weights to bf16
  cvt_f32_to_bf16<<<gEl, blk, 0, stream>>>(x,  xbf,  n);
  cvt_f32_to_bf16<<<gEl, blk, 0, stream>>>(Wq, wqbf, n);
  cvt_f32_to_bf16<<<gEl, blk, 0, stream>>>(Wk, wkbf, n);
  cvt_f32_to_bf16<<<gEl, blk, 0, stream>>>(Wv, wvbf, n);
  cvt_f32_to_bf16<<<gEl, blk, 0, stream>>>(Wo, wobf, n);

  // 2) projections (WMMA)
  gemm_bf16_nt<<<gGemm, blk, 0, stream>>>(xbf, wqbf, qf, T_SEQ, EMB, EMB);
  gemm_bf16_nt<<<gGemm, blk, 0, stream>>>(xbf, wkbf, kf, T_SEQ, EMB, EMB);
  gemm_bf16_nt<<<gGemm, blk, 0, stream>>>(xbf, wvbf, vf, T_SEQ, EMB, EMB);

  // 3) RoPE (+ head-major bf16), V transpose, lambda scalar
  rope_cvt<<<gEl, blk, 0, stream>>>(qf, cosv, sinv, qbf, 0.125f); // hd^-0.5
  rope_cvt<<<gEl, blk, 0, stream>>>(kf, cosv, sinv, kbf, 1.0f);
  v_cvt_tr<<<gEl, blk, 0, stream>>>(vf, vtbf);
  lambda_k<<<1, 32, 0, stream>>>(lq1, lk1, lq2, lk2, lam);

  // 4) differential flash attention (WMMA)
  diff_flash<<<dim3(T_SEQ / 64, NH), blk, 0, stream>>>(qbf, kbf, vtbf, lam,
                                                       subl, attnbf);

  // 5) output projection (WMMA) -> f32 d_out
  gemm_bf16_nt<<<gGemm, blk, 0, stream>>>(attnbf, wobf, out, T_SEQ, EMB, EMB);
}
